// WindowAttention_55327768708008
// MI455X (gfx1250) — matrix operand order repaired and checked by reading, not compiled, not run on hardware
//
#include <hip/hip_runtime.h>
#include <hip/hip_bf16.h>
#include <math.h>

// ---------------------------------------------------------------------------
// Fused Swin window attention for gfx1250 (MI455X), wave32 + WMMA + TDM.
// One workgroup (256 threads = 8 wave32) per window. All intermediates in LDS
// with bank-conflict-free padded strides. GEMMs: v_wmma_f32_16x16x32_f16.
// Each wave register-blocks one M-tile (A fragments cached in VGPRs) and
// streams B fragments. V is stored transposed so the P@V B-fragment loads and
// the V scatter are contiguous. wave index goes through readfirstlane so all
// tile control flow is scalar (SGPR loop counters, s_cbranch scatter select).
// Prep kernels pre-convert weights to f16 and fuse rel-pos bias + window mask
// into one f16 table (L2-resident). Stage-1 x tile is staged into LDS by the
// Tensor Data Mover when available.
// ---------------------------------------------------------------------------

typedef __attribute__((ext_vector_type(16))) _Float16     v16h;
typedef __attribute__((ext_vector_type(8)))  float        v8f;
typedef __attribute__((ext_vector_type(4)))  unsigned int u32x4;
typedef __attribute__((ext_vector_type(8)))  int          i32x8;
typedef __attribute__((ext_vector_type(4)))  int          i32x4;

#define NWIN   4096
#define NTOK   49
#define NPAD   64          // 49 padded to 4x16 tiles
#define CDIM   256
#define NHEAD  8
#define HDIM   32
#define NMASK  64
#define QKVC   768         // 3*C
#define SCALE  0.17677669529663687f   // 1/sqrt(32)

// Padded LDS strides (halves) -> conflict-free bank steps (64 banks x 4B):
#define X_STR   264        // 528B  = bank step 4  (x16 lanes: all distinct)
#define QK_STR  36         // 72B   = bank step 18
#define VT_STR  68         // 136B  = bank step 34
#define PS_STR  68
#define OS_STR  264

// LDS halves layout (total 122880 halves = 245760 B <= 320KB/WGP)
#define XS_OFF   0                               // [64][264]      x (f16)
#define QS_OFF   (XS_OFF + NPAD*X_STR)           // [8][64][36]    q*scale
#define KS_OFF   (QS_OFF + NHEAD*NPAD*QK_STR)    // [8][64][36]    k
#define VS_OFF   (KS_OFF + NHEAD*NPAD*QK_STR)    // [8][32][68]    v^T
#define PS_OFF   (VS_OFF + NHEAD*HDIM*VT_STR)    // [8][64][68]    scores/probs (+TDM stage)
#define OS_OFF   (PS_OFF + NHEAD*NPAD*PS_STR)    // [64][264]      attn output
#define LDS_HALVES (OS_OFF + NPAD*OS_STR)
#define LDS_BYTES  (LDS_HALVES * 2)

// Workspace layout (bytes)
#define WS_QKVW16   0                         // 768*256 f16 = 393216 B
#define WS_PROJW16  393216                    // 256*256 f16 = 131072 B
#define WS_BM16     524288                    // 64*8*49*49 f16 = 2458624 B
#define BM_ELEMS    (NMASK * NHEAD * NTOK * NTOK)

// --- A fragment (16x32 f16, MxK): lane = M row (0..15) per half-wave group;
//     halves 0..15 -> K = g*16..g*16+15 (g = lane>>4)
__device__ __forceinline__ v16h load_a_lds(const _Float16* base, int stride,
                                           int m0, int k0, int lane) {
  const int m  = m0 + (lane & 15);
  const int kh = (lane >> 4) * 16;
  const _Float16* p = base + m * stride + k0;
  v16h a;
#pragma unroll
  for (int j = 0; j < 8; ++j)  a[j]     = p[kh + j];
#pragma unroll
  for (int j = 0; j < 8; ++j)  a[8 + j] = p[kh + 8 + j];
  return a;
}

// --- B fragment (32x16 f16, KxN) from a row-major [N][K] f16 buffer
//     (B = W^T with W row-major): lane -> column n, 16 sequential K.
__device__ __forceinline__ v16h load_bt_f16(const _Float16* __restrict__ base,
                                            int stride, int n0, int k0, int lane) {
  const int n  = n0 + (lane & 15);
  const int kb = k0 + (lane >> 4) * 16;
  const _Float16* p = base + n * stride + kb;
  v16h b;
#pragma unroll
  for (int j = 0; j < 16; ++j) b[j] = p[j];
  return b;
}

__device__ __forceinline__ v8f wmma_f16(v16h a, v16h b, v8f c) {
  return __builtin_amdgcn_wmma_f32_16x16x32_f16(
      /*neg_a=*/false, a, /*neg_b=*/false, b,
      /*c_mod=*/(short)0, c, /*reuse_a=*/false, /*reuse_b=*/false);
}

// ---------------------------------------------------------------------------
// Prep kernel 1: weights f32 -> f16 (one-time, L2-resident afterwards).
// ---------------------------------------------------------------------------
__global__ void prep_weights_kernel(const float* __restrict__ qkv_w,
                                    const float* __restrict__ proj_w,
                                    _Float16* __restrict__ qkv16,
                                    _Float16* __restrict__ proj16) {
  const int i = blockIdx.x * 256 + threadIdx.x;
  if (i < QKVC * CDIM)  qkv16[i]  = (_Float16)qkv_w[i];
  if (i < CDIM * CDIM)  proj16[i] = (_Float16)proj_w[i];
}

// ---------------------------------------------------------------------------
// Prep kernel 2: fused bias table bm[mw][h][i][j] = rpb[rel[i,j], h] + mask.
// ---------------------------------------------------------------------------
__global__ void prep_bias_kernel(const float* __restrict__ mask,
                                 const float* __restrict__ rpb_table,
                                 const int*   __restrict__ rel_index,
                                 _Float16* __restrict__ bm) {
  const int idx = blockIdx.x * 256 + threadIdx.x;
  if (idx >= BM_ELEMS) return;
  const int j  = idx % NTOK;
  int t        = idx / NTOK;
  const int i  = t % NTOK;  t /= NTOK;
  const int h  = t % NHEAD;
  const int mw = t / NHEAD;
  const float v = rpb_table[rel_index[i * NTOK + j] * NHEAD + h]
                + mask[(size_t)mw * NTOK * NTOK + i * NTOK + j];
  bm[idx] = (_Float16)v;
}

// ---------------------------------------------------------------------------
// Main fused kernel.
// ---------------------------------------------------------------------------
__global__ void __launch_bounds__(256, 1)
swin_window_attn_kernel(const float*    __restrict__ x,
                        const _Float16* __restrict__ qkv_w16,
                        const float*    __restrict__ qkv_b,
                        const _Float16* __restrict__ proj_w16,
                        const float*    __restrict__ proj_b,
                        const _Float16* __restrict__ bm,
                        float*          __restrict__ out) {
  extern __shared__ __align__(16) _Float16 smem[];
  _Float16* Xs = smem + XS_OFF;
  _Float16* Qs = smem + QS_OFF;
  _Float16* Ks = smem + KS_OFF;
  _Float16* Vs = smem + VS_OFF;     // transposed: [h][d][token]
  _Float16* Ps = smem + PS_OFF;
  _Float16* Os = smem + OS_OFF;

  const int b    = blockIdx.x;
  const int tid  = threadIdx.x;
  // Wave index is uniform by construction; readfirstlane pins it (and all
  // derived tile control flow) into SGPRs -> scalar branches, SALU counters.
  const int wave = __builtin_amdgcn_readfirstlane(tid >> 5);
  const int lane = tid & 31;

  // -------- Stage 1: x[b] (49x256 f32) -> LDS f16, rows padded with zeros ---
  const float* xb = x + (size_t)b * NTOK * CDIM;
#if __has_builtin(__builtin_amdgcn_tensor_load_to_lds) && \
    __has_builtin(__builtin_amdgcn_s_wait_tensorcnt)
  // Tensor Data Mover: DMA the whole 49x256 f32 tile into the (still unused)
  // Ps region, then convert to f16. 2D descriptor, data_size=4B.
  {
    if (wave == 0) {
      const unsigned long long ga = (unsigned long long)(uintptr_t)xb;
      // D# group0: count=1 | lds_addr[63:32] | global_addr[120:64] | type=2
      u32x4 g0 = { 1u,
                   (unsigned)(PS_OFF * 2),
                   (unsigned)(ga & 0xFFFFFFFFull),
                   (unsigned)((ga >> 32) & 0x01FFFFFFull) | (2u << 30) };
      // D# group1: data_size=2 (4B); tensor_dim0=256, tensor_dim1=49;
      //            tile_dim0=256, tile_dim1=49; tensor_dim0_stride=256.
      i32x8 g1 = { (int)0x20000,
                   (int)(256u << 16),
                   (int)(49u  << 16),
                   (int)(256u << 16),
                   (int)49,
                   (int)256,
                   0, 0 };
      i32x4 g2 = { 0, 0, 0, 0 };
      i32x4 g3 = { 0, 0, 0, 0 };
#if __clang_major__ >= 23
      i32x8 g4 = { 0, 0, 0, 0, 0, 0, 0, 0 };
      __builtin_amdgcn_tensor_load_to_lds(g0, g1, g2, g3, g4, 0);
#else
      __builtin_amdgcn_tensor_load_to_lds(g0, g1, g2, g3, 0);
#endif
      __builtin_amdgcn_s_wait_tensorcnt(0);
    }
    __syncthreads();
    const float* Xstage = (const float*)(const void*)Ps;
    for (int i = tid; i < NPAD * CDIM; i += 256) {
      const int r = i >> 8, c = i & 255;
      Xs[r * X_STR + c] = (_Float16)((r < NTOK) ? Xstage[r * CDIM + c] : 0.0f);
    }
  }
#else
  for (int i = tid; i < NPAD * CDIM; i += 256) {
    const int r = i >> 8, c = i & 255;
    Xs[r * X_STR + c] = (_Float16)((r < NTOK) ? xb[r * CDIM + c] : 0.0f);
  }
#endif
  __syncthreads();

  // -------- Stage 2: QKV = Xs @ qkv_w^T + qkv_b ; scatter to Q/K/V^T heads --
  // Each wave owns one M-tile (A fragments cached in 64 VGPRs) and streams
  // 24 of the 48 N-tiles of the 768-wide QKV output.
  {
    const int mt = wave & 3;
    const int m0 = mt * 16;
    v16h afr[8];
#pragma unroll
    for (int kk = 0; kk < 8; ++kk)
      afr[kk] = load_a_lds(Xs, X_STR, m0, kk * 32, lane);

    const int ntBeg = (wave >> 2) * 24;
    for (int nt = ntBeg; nt < ntBeg + 24; ++nt) {
      const int n0 = nt * 16;                // scalar (SGPR)
      __builtin_prefetch(qkv_w16 + (n0 + 16 + (lane & 15)) * CDIM, 0, 1);
      v8f acc = {};
#pragma unroll
      for (int kk = 0; kk < 8; ++kk) {
        const v16h bw = load_bt_f16(qkv_w16, CDIM, n0, kk * 32, lane);
        acc = wmma_f16(afr[kk], bw, acc);
      }
      const int col  = n0 + (lane & 15);     // [0, 768)
      const int part = n0 >> 8;              // scalar: 0=q 1=k 2=v
      const int cc   = col & 255;
      const int hh   = cc >> 5;
      const int dd   = cc & 31;
      const float bias = qkv_b[col];
      if (part == 0) {
#pragma unroll
        for (int r = 0; r < 8; ++r) {
          const int row = m0 + ((lane >> 4) << 3) + r;
          Qs[(hh * NPAD + row) * QK_STR + dd] = (_Float16)((acc[r] + bias) * SCALE);
        }
      } else if (part == 1) {
#pragma unroll
        for (int r = 0; r < 8; ++r) {
          const int row = m0 + ((lane >> 4) << 3) + r;
          Ks[(hh * NPAD + row) * QK_STR + dd] = (_Float16)(acc[r] + bias);
        }
      } else {   // V stored transposed: contiguous in r
        _Float16* vrow = Vs + (hh * HDIM + dd) * VT_STR + m0 + ((lane >> 4) << 3);
#pragma unroll
        for (int r = 0; r < 8; ++r) vrow[r] = (_Float16)(acc[r] + bias);
      }
    }
  }
  __syncthreads();

  // -------- Stage 3: per-head attention; wave h owns head h -----------------
  {
    const int h = wave;
    const _Float16* Qh = Qs + h * NPAD * QK_STR;
    const _Float16* Kh = Ks + h * NPAD * QK_STR;
    const _Float16* Vt = Vs + h * HDIM * VT_STR;   // [d][token]
    _Float16*       Ph = Ps + h * NPAD * PS_STR;
    const _Float16* bmh = bm + ((size_t)(b % NMASK) * NHEAD + h) * NTOK * NTOK;

    // S = (q*scale) @ k^T + fused(bias+mask)   (4x4 tiles, K=32)
    v16h bk[4];
#pragma unroll
    for (int nt = 0; nt < 4; ++nt)
      bk[nt] = load_bt_f16(Kh, QK_STR, nt * 16, 0, lane);   // B = K^T
#pragma unroll
    for (int mt = 0; mt < 4; ++mt) {
      const v16h a = load_a_lds(Qh, QK_STR, mt * 16, 0, lane);
#pragma unroll
      for (int nt = 0; nt < 4; ++nt) {
        v8f s = {};
        s = wmma_f16(a, bk[nt], s);
#pragma unroll
        for (int r = 0; r < 8; ++r) {
          const int i = mt * 16 + ((lane >> 4) << 3) + r;
          const int j = nt * 16 + (lane & 15);
          float val = s[r];
          if (j >= NTOK) {
            val = -30000.0f;                           // pad columns: -inf-ish
          } else if (i < NTOK) {
            val += (float)bmh[i * NTOK + j];
          }
          Ph[i * PS_STR + j] = (_Float16)val;
        }
      }
    }
    __syncthreads();

    // Row softmax (each lane handles 2 rows of its head's 64x64 score tile).
    for (int rr = lane; rr < NPAD; rr += 32) {
      _Float16* row = Ph + rr * PS_STR;
      float mx = -3.4e38f;
#pragma unroll
      for (int j = 0; j < NPAD; ++j) mx = fmaxf(mx, (float)row[j]);
      float sum = 0.0f;
#pragma unroll
      for (int j = 0; j < NPAD; ++j) sum += __expf((float)row[j] - mx);
      const float inv = 1.0f / sum;
#pragma unroll
      for (int j = 0; j < NPAD; ++j)
        row[j] = (_Float16)(__expf((float)row[j] - mx) * inv);
    }
    __syncthreads();

    // O_h = P @ V   (4 M-tiles x 2 N-tiles, K=64 -> 2 WMMA each).
    // V^T layout makes every B fragment 16 contiguous halves per lane.
    v16h bv[2][2];
#pragma unroll
    for (int nt = 0; nt < 2; ++nt)
#pragma unroll
      for (int kk = 0; kk < 2; ++kk)
        bv[nt][kk] = load_bt_f16(Vt, VT_STR, nt * 16, kk * 32, lane);
#pragma unroll
    for (int mt = 0; mt < 4; ++mt) {
      v16h aP[2];
#pragma unroll
      for (int kk = 0; kk < 2; ++kk)
        aP[kk] = load_a_lds(Ph, PS_STR, mt * 16, kk * 32, lane);
#pragma unroll
      for (int nt = 0; nt < 2; ++nt) {
        v8f acc = {};
#pragma unroll
        for (int kk = 0; kk < 2; ++kk)
          acc = wmma_f16(aP[kk], bv[nt][kk], acc);
#pragma unroll
        for (int r = 0; r < 8; ++r) {
          const int i = mt * 16 + ((lane >> 4) << 3) + r;
          const int d = nt * 16 + (lane & 15);
          Os[i * OS_STR + h * HDIM + d] = (_Float16)acc[r];
        }
      }
    }
  }
  __syncthreads();

  // -------- Stage 4: out = Os @ proj_w^T + proj_b ---------------------------
  // Each wave owns one M-tile (8 cached A fragments), streams 8 N-tiles.
  float* ob = out + (size_t)b * NTOK * CDIM;
  {
    const int mt = wave & 3;
    const int m0 = mt * 16;
    v16h ao[8];
#pragma unroll
    for (int kk = 0; kk < 8; ++kk)
      ao[kk] = load_a_lds(Os, OS_STR, m0, kk * 32, lane);

    const int ntBeg = (wave >> 2) * 8;
    for (int nt = ntBeg; nt < ntBeg + 8; ++nt) {
      const int n0 = nt * 16;
      __builtin_prefetch(proj_w16 + (n0 + 16 + (lane & 15)) * CDIM, 0, 1);
      v8f acc = {};
#pragma unroll
      for (int kk = 0; kk < 8; ++kk) {
        const v16h bw = load_bt_f16(proj_w16, CDIM, n0, kk * 32, lane);
        acc = wmma_f16(ao[kk], bw, acc);
      }
      const int col = n0 + (lane & 15);
      const float pb = proj_b[col];
#pragma unroll
      for (int r = 0; r < 8; ++r) {
        const int row = m0 + ((lane >> 4) << 3) + r;
        if (row < NTOK) ob[row * CDIM + col] = acc[r] + pb;
      }
    }
  }
}

extern "C" void kernel_launch(void* const* d_in, const int* in_sizes, int n_in,
                              void* d_out, int out_size, void* d_ws, size_t ws_size,
                              hipStream_t stream) {
  (void)in_sizes; (void)n_in; (void)ws_size; (void)out_size;
  const float* x       = (const float*)d_in[0];
  const float* mask    = (const float*)d_in[1];
  const float* qkv_w   = (const float*)d_in[2];
  const float* qkv_b   = (const float*)d_in[3];
  const float* proj_w  = (const float*)d_in[4];
  const float* proj_b  = (const float*)d_in[5];
  const float* rpb     = (const float*)d_in[6];
  const int*   rel     = (const int*)d_in[7];
  float* out           = (float*)d_out;

  _Float16* qkv16  = (_Float16*)((char*)d_ws + WS_QKVW16);
  _Float16* proj16 = (_Float16*)((char*)d_ws + WS_PROJW16);
  _Float16* bm16   = (_Float16*)((char*)d_ws + WS_BM16);

  // One-time (per call) prep: f16 weights + fused bias/mask table.
  prep_weights_kernel<<<dim3((QKVC * CDIM + 255) / 256), dim3(256), 0, stream>>>(
      qkv_w, proj_w, qkv16, proj16);
  prep_bias_kernel<<<dim3((BM_ELEMS + 255) / 256), dim3(256), 0, stream>>>(
      mask, rpb, rel, bm16);

  swin_window_attn_kernel<<<dim3(NWIN), dim3(256), LDS_BYTES, stream>>>(
      x, qkv16, qkv_b, proj16, proj_b, bm16, out);
}
